// TypedTreeCell_5557687681542
// MI455X (gfx1250) — compile-verified, matrix operand order checked by reading, not run on hardware
//
#include <hip/hip_runtime.h>
#include <hip/hip_bf16.h>
#include <stdint.h>

#define N_NODES 100000
#define KCH 8
#define XS 128
#define HS 128
#define NT 8

typedef __attribute__((ext_vector_type(16))) __bf16 v16bf;
typedef __attribute__((ext_vector_type(8)))  float  v8f;

union BFrag { v16bf v; uint4 u[2]; };

__device__ __forceinline__ unsigned short f2bf(float f) {
  unsigned int u = __float_as_uint(f);
  u += 0x7FFFu + ((u >> 16) & 1u);        // round-to-nearest-even
  return (unsigned short)(u >> 16);
}
__device__ __forceinline__ void store_bf4(unsigned short* d, float4 v) {
  uint2 p;
  p.x = (unsigned)f2bf(v.x) | ((unsigned)f2bf(v.y) << 16);
  p.y = (unsigned)f2bf(v.z) | ((unsigned)f2bf(v.w) << 16);
  *(uint2*)d = p;
}
__device__ __forceinline__ float sigm(float x) { return 1.0f / (1.0f + __expf(-x)); }

// A/B bf16 fragment for V_WMMA_F32_16X16X32_BF16 (ISA 7.12.2):
// lane<16 : row/col = lane,    K = k0..k0+7 (v0-3), k0+16..k0+23 (v4-7)
// lane>=16: row/col = lane-16, K = k0+8..k0+15,     k0+24..k0+31
__device__ __forceinline__ v16bf frag_lds(const unsigned short* base, int row, int rs,
                                          int k0, int lane) {
  int half = lane >> 4;
  const unsigned short* p = base + row * rs + k0 + 8 * half;
  BFrag f;
  f.u[0] = *(const uint4*)p;
  f.u[1] = *(const uint4*)(p + 16);
  return f.v;
}
__device__ __forceinline__ v16bf frag_glb(const unsigned short* rowp, int k0, int lane) {
  int half = lane >> 4;
  BFrag f;
  f.u[0] = *(const uint4*)(rowp + k0 + 8 * half);
  f.u[1] = *(const uint4*)(rowp + k0 + 16 + 8 * half);
  return f.v;
}

// ---------------- pass 0: zero per-type counters ----------------
__global__ void k_zero(int* counters) {
  if (threadIdx.x < NT) counters[threadIdx.x] = 0;
}

// ---------------- pass 1: bucket nodes by type ----------------
__global__ void k_bucket(const int* __restrict__ type_id, int* counters,
                         int* __restrict__ buckets) {
  int i = blockIdx.x * 256 + threadIdx.x;
  if (i < N_NODES) {
    int t = type_id[i] & (NT - 1);
    int pos = atomicAdd(&counters[t], 1);
    buckets[t * N_NODES + pos] = i;
  }
}

// ---------------- pass 2: weights -> bf16, k-contiguous (transposed) ------
// WU [t][n:384][k:256], k<128 from W_iou[t][k][n], k>=128 from U_iou[t][k-128][n]
// UfT[t][n:128][k:128] = U_f[t][k][n]
__global__ void k_convw(const float* __restrict__ W_iou, const float* __restrict__ U_iou,
                        const float* __restrict__ U_f,
                        unsigned short* __restrict__ WU, unsigned short* __restrict__ UfT) {
  const int T1 = NT * 384 * 256;
  const int T2 = NT * 128 * 128;
  int idx = blockIdx.x * 256 + threadIdx.x;
  if (idx < T1) {
    int t = idx / (384 * 256);
    int r = idx % (384 * 256);
    int n = r >> 8;
    int k = r & 255;
    float v = (k < 128) ? W_iou[(size_t)t * 128 * 384 + k * 384 + n]
                        : U_iou[(size_t)t * 128 * 384 + (k - 128) * 384 + n];
    WU[idx] = f2bf(v);
  } else if (idx < T1 + T2) {
    int j = idx - T1;
    int t = j >> 14;
    int r = j & 16383;
    int n = r >> 7;
    int k = r & 127;
    UfT[j] = f2bf(U_f[(size_t)t * 128 * 128 + k * 128 + n]);
  }
}

// ---------------- pass 3: 16 same-type nodes per block, WMMA everything ---
// LDS layout (dynamic, 76096 B):
//   piou  f32 [16][384]          @ 0      (24576 B)
//   cagg  f32 [16][128]          @ 24576  ( 8192 B)
//   nh    bf16[128][136]         @ 32768  (34816 B)  rows = node*8+child
//   acat  bf16[16][264]          @ 67584  ( 8448 B)  [ x | h_sum ]
//   nidx  int [16]               @ 76032
__global__ __launch_bounds__(128) void k_tree(
    const float* __restrict__ x, const float* __restrict__ n_h,
    const float* __restrict__ n_c, const float* __restrict__ b_iou,
    const float* __restrict__ b_f, const int* __restrict__ counters,
    const int* __restrict__ buckets, const unsigned short* __restrict__ WU,
    const unsigned short* __restrict__ UfT,
    float* __restrict__ h_out, float* __restrict__ c_out) {
  const int t = blockIdx.y;
  const int cnt = counters[t];
  const int base = blockIdx.x * 16;
  if (base >= cnt) return;
  const int m = min(16, cnt - base);

  extern __shared__ char smem[];
  float* piou          = (float*)smem;
  float* cagg          = (float*)(smem + 24576);
  unsigned short* nh   = (unsigned short*)(smem + 32768);
  unsigned short* acat = (unsigned short*)(smem + 67584);
  int* nidx            = (int*)(smem + 76032);

  const int tid  = threadIdx.x;
  const int lane = tid & 31;
  const int wv   = tid >> 5;
  const int r16  = lane & 15;
  const int half = lane >> 4;

  if (tid < 16) nidx[tid] = buckets[t * N_NODES + min(base + tid, cnt - 1)];
  __syncthreads();

  // --- stage 1: gather & bf16-convert tiles ---
  // n_h tile: 16 nodes * 8 children * 128 = 4096 float4
  for (int i = tid; i < 4096; i += 128) {
    int nl = i >> 8;                 // node local
    int rem = i & 255;               // float4 within node
    float4 v = *(const float4*)(n_h + (size_t)nidx[nl] * (KCH * HS) + rem * 4);
    int r = nl * 8 + (rem >> 5);     // row = node*8+child
    int k = (rem & 31) * 4;
    store_bf4(nh + r * 136 + k, v);
  }
  // x tile: 16*128 = 512 float4
  for (int i = tid; i < 512; i += 128) {
    int nl = i >> 5;
    int k = (i & 31) * 4;
    float4 v = *(const float4*)(x + (size_t)nidx[nl] * XS + k);
    store_bf4(acat + nl * 264 + k, v);
  }
  // h_sum (f32 accumulate over 8 children, cache hits): 512 float4
  for (int i = tid; i < 512; i += 128) {
    int nl = i >> 5;
    int k = (i & 31) * 4;
    const float* p = n_h + (size_t)nidx[nl] * (KCH * HS) + k;
    float4 s = make_float4(0.f, 0.f, 0.f, 0.f);
#pragma unroll
    for (int ch = 0; ch < KCH; ++ch) {
      float4 v = *(const float4*)(p + ch * HS);
      s.x += v.x; s.y += v.y; s.z += v.z; s.w += v.w;
    }
    store_bf4(acat + nl * 264 + 128 + k, s);
  }
  __syncthreads();

  // --- stage 2: [x|h_sum](16x256) @ [W_iou;U_iou]^T -> piou 16x384 ---
  const unsigned short* wu_t = WU + (size_t)t * 384 * 256;
  for (int j = 0; j < 6; ++j) {
    int ct = wv * 6 + j;                                   // 24 col tiles / 4 waves
    const unsigned short* brow = wu_t + (size_t)(ct * 16 + r16) * 256;
    v8f acc = {};
#pragma unroll
    for (int kt = 0; kt < 8; ++kt) {
      v16bf a = frag_lds(acat, r16, 264, kt * 32, lane);
      v16bf b = frag_glb(brow, kt * 32, lane);
      acc = __builtin_amdgcn_wmma_f32_16x16x32_bf16(false, a, false, b,
                                                    (short)0, acc, false, false);
    }
#pragma unroll
    for (int v = 0; v < 8; ++v)
      piou[(v + 8 * half) * 384 + ct * 16 + r16] = acc[v];
  }

  // --- stage 3: f-gate GEMM (128x128) fused with sigmoid * n_c child-sum ---
  // D tile rows = 2 nodes x 8 children -> each lane reduces a full child set.
  const unsigned short* uf_t = UfT + (size_t)t * 128 * 128;
  for (int tt = 0; tt < 2; ++tt) {
    int tr = wv * 2 + tt;                                  // 8 row tiles / 4 waves
    int nl = 2 * tr + half;                                // node local for this lane
    const float* ncp = n_c + (size_t)nidx[nl] * (KCH * HS);
    for (int tc = 0; tc < 8; ++tc) {
      const unsigned short* brow = uf_t + (size_t)(tc * 16 + r16) * 128;
      v8f acc = {};
#pragma unroll
      for (int kt = 0; kt < 4; ++kt) {
        v16bf a = frag_lds(nh, tr * 16 + r16, 136, kt * 32, lane);
        v16bf b = frag_glb(brow, kt * 32, lane);
        acc = __builtin_amdgcn_wmma_f32_16x16x32_bf16(false, a, false, b,
                                                      (short)0, acc, false, false);
      }
      int h = tc * 16 + r16;
      float bf = b_f[t * HS + h];
      float cs = 0.f;
#pragma unroll
      for (int v = 0; v < 8; ++v) {                        // child = v (D-layout)
        cs += sigm(acc[v] + bf) * ncp[v * HS + h];
      }
      cagg[nl * HS + h] = cs;
    }
  }
  __syncthreads();

  // --- stage 4: gates + outputs ---
  for (int i = tid; i < 16 * HS; i += 128) {
    int nl = i >> 7;
    int h = i & 127;
    if (nl < m) {
      float ig = piou[nl * 384 + h]       + b_iou[t * 384 + h];
      float og = piou[nl * 384 + 128 + h] + b_iou[t * 384 + 128 + h];
      float ug = piou[nl * 384 + 256 + h] + b_iou[t * 384 + 256 + h];
      float c  = sigm(ig) * tanhf(ug) + cagg[nl * HS + h];
      float hh = sigm(og) * tanhf(c);
      size_t o = (size_t)nidx[nl] * HS + h;
      h_out[o] = hh;
      c_out[o] = c;
    }
  }
}

extern "C" void kernel_launch(void* const* d_in, const int* in_sizes, int n_in,
                              void* d_out, int out_size, void* d_ws, size_t ws_size,
                              hipStream_t stream) {
  const float* x      = (const float*)d_in[0];
  const float* n_h    = (const float*)d_in[1];
  const float* n_c    = (const float*)d_in[2];
  const int*   typ    = (const int*)d_in[3];
  const float* W_iou  = (const float*)d_in[4];
  const float* U_iou  = (const float*)d_in[5];
  const float* b_iou  = (const float*)d_in[6];
  const float* U_f    = (const float*)d_in[7];
  const float* b_f    = (const float*)d_in[8];
  float* out = (float*)d_out;

  // workspace layout (all 16B aligned)
  char* ws = (char*)d_ws;
  int* counters          = (int*)ws;                           // 32 B
  int* buckets           = (int*)(ws + 64);                    // 8*N*4 = 3,200,000
  unsigned short* WU     = (unsigned short*)(ws + 3200064);    // 1,572,864
  unsigned short* UfT    = (unsigned short*)(ws + 4772928);    // 262,144

  k_zero<<<1, 32, 0, stream>>>(counters);
  k_bucket<<<(N_NODES + 255) / 256, 256, 0, stream>>>(typ, counters, buckets);
  k_convw<<<(NT * 384 * 256 + NT * 128 * 128 + 255) / 256, 256, 0, stream>>>(
      W_iou, U_iou, U_f, WU, UfT);

  dim3 grid((N_NODES + 15) / 16, NT);
  k_tree<<<grid, 128, 76096, stream>>>(x, n_h, n_c, b_iou, b_f, counters, buckets,
                                       WU, UfT, out, out + (size_t)N_NODES * HS);
}
